// VariationalGCNEncoder_72438918414913
// MI455X (gfx1250) — matrix operand order repaired and checked by reading, not compiled, not run on hardware
//
#include <hip/hip_runtime.h>

#define NN   10000
#define EE   160000
#define CIN  1024
#define HH   1024     // 2*OUT
#define COUT 512
#define NEG  0.1f
#define EPSV 1e-5f

typedef unsigned int  u32x4 __attribute__((ext_vector_type(4)));
typedef __attribute__((ext_vector_type(16))) __bf16 bf16x16;
typedef __attribute__((ext_vector_type(8)))  float  f32x8;

struct FragPair { u32x4 a, b; };   // 32 bytes -> bit_cast to bf16x16

// ---- CDNA5 async global->LDS copy (ASYNCcnt path), with portable fallback ----
#if defined(__has_builtin)
#if __has_builtin(__builtin_amdgcn_global_load_async_to_lds_b128) && \
    __has_builtin(__builtin_amdgcn_s_wait_asynccnt)
#define USE_ASYNC_LDS 1
#endif
#endif

#ifdef USE_ASYNC_LDS
typedef int i32x4v __attribute__((ext_vector_type(4)));
typedef __attribute__((address_space(1))) i32x4v gas_i4;   // HIP "__device__" AS
typedef __attribute__((address_space(3))) i32x4v las_i4;   // HIP "__shared__" AS
__device__ __forceinline__ void cp16(const unsigned short* g, unsigned short* l) {
  __builtin_amdgcn_global_load_async_to_lds_b128((gas_i4*)g, (las_i4*)l, 0, 0);
}
template <int Ncnt> __device__ __forceinline__ void cp_wait_n() {
  __builtin_amdgcn_s_wait_asynccnt(Ncnt);
}
#else
__device__ __forceinline__ void cp16(const unsigned short* g, unsigned short* l) {
  *(u32x4*)l = *(const u32x4*)g;
}
template <int Ncnt> __device__ __forceinline__ void cp_wait_n() {}
#endif

__device__ __forceinline__ unsigned short f2bf(float f) {
  unsigned int u = __float_as_uint(f);
  u += 0x7fffu + ((u >> 16) & 1u);           // round-to-nearest-even
  return (unsigned short)(u >> 16);
}
__device__ __forceinline__ float bf2f(unsigned short h) {
  return __uint_as_float(((unsigned int)h) << 16);
}
__device__ __forceinline__ void atomAddF(float* p, float v) { unsafeAtomicAdd(p, v); }

// ======================= GraphNorm stats =======================
__global__ void k_stats_zero(float* s, float* s2) {
  int c = blockIdx.x * blockDim.x + threadIdx.x;
  s[c] = 0.f; s2[c] = 0.f;
}

__global__ void k_stats_partial(const float* __restrict__ x,
                                float* __restrict__ s, float* __restrict__ s2,
                                int rowsPer) {
  int c  = blockIdx.x * blockDim.x + threadIdx.x;
  int r0 = blockIdx.y * rowsPer;
  int r1 = r0 + rowsPer; if (r1 > NN) r1 = NN;
  float a = 0.f, b = 0.f;
  for (int n = r0; n < r1; ++n) {
    float v = x[(size_t)n * CIN + c];
    a += v; b += v * v;
  }
  atomAddF(&s[c], a);
  atomAddF(&s2[c], b);
}

__global__ void k_stats_final(const float* __restrict__ s, const float* __restrict__ s2,
                              const float* __restrict__ msc, const float* __restrict__ gw,
                              const float* __restrict__ gb,
                              float* __restrict__ alpha, float* __restrict__ beta) {
  int c = blockIdx.x * blockDim.x + threadIdx.x;
  float mean = s[c] * (1.f / NN);
  float ms   = msc[c];
  // var = E[x^2] - 2*ms*mean*E[x] + ms^2*mean^2
  float var  = s2[c] * (1.f / NN) - (2.f * ms - ms * ms) * mean * mean;
  float a    = rsqrtf(var + EPSV) * gw[c];
  alpha[c] = a;
  beta[c]  = gb[c] - ms * mean * a;
}

// x_norm = x*alpha[c] + beta[c], split into bf16 hi+lo (linear row-major layout)
__global__ void k_norm_split(const float* __restrict__ x,
                             const float* __restrict__ alpha, const float* __restrict__ beta,
                             unsigned short* __restrict__ hi, unsigned short* __restrict__ lo) {
  size_t i = (size_t)blockIdx.x * blockDim.x + threadIdx.x;
  int c = (int)(i & (CIN - 1));
  float v = x[i] * alpha[c] + beta[c];
  unsigned short h = f2bf(v);
  hi[i] = h;
  lo[i] = f2bf(v - bf2f(h));
}

// ======================= weight prep: WMMA-fragment tiling =======================
// B-fragment layout for 16x16x32 bf16 WMMA (wave32):
//   tiled[((kt*ntiles + nt)*32 + lane)*16 + i] = B[kt*32 + (i>>3)*16 + (lane>>4)*8 + (i&7)][nt*16 + (lane&15)]
// ntiles = Nc/16 = 64 for both GEMMs. Per-lane fragment = 32 contiguous bytes.
__global__ void k_tile_w1(const float* __restrict__ W,
                          unsigned short* __restrict__ hi, unsigned short* __restrict__ lo) {
  size_t t = (size_t)blockIdx.x * blockDim.x + threadIdx.x;  // over CIN*HH
  int i    = (int)(t & 15);
  int lane = (int)((t >> 4) & 31);
  int rest = (int)(t >> 9);
  int nt   = rest & 63;
  int kt   = rest >> 6;
  int k = kt * 32 + (i >> 3) * 16 + (lane >> 4) * 8 + (i & 7);
  int n = nt * 16 + (lane & 15);
  float v = W[(size_t)k * HH + n];
  unsigned short h = f2bf(v);
  hi[t] = h;
  lo[t] = f2bf(v - bf2f(h));
}

// pack [Wm | Ws] -> [HH x 1024] and tile, split bf16
__global__ void k_tile_wcat(const float* __restrict__ Wm, const float* __restrict__ Ws,
                            unsigned short* __restrict__ hi, unsigned short* __restrict__ lo) {
  size_t t = (size_t)blockIdx.x * blockDim.x + threadIdx.x;  // over HH*1024
  int i    = (int)(t & 15);
  int lane = (int)((t >> 4) & 31);
  int rest = (int)(t >> 9);
  int nt   = rest & 63;
  int kt   = rest >> 6;
  int k = kt * 32 + (i >> 3) * 16 + (lane >> 4) * 8 + (i & 7);
  int n = nt * 16 + (lane & 15);
  float v = (n < COUT) ? Wm[(size_t)k * COUT + n] : Ws[(size_t)k * COUT + (n - COUT)];
  unsigned short h = f2bf(v);
  hi[t] = h;
  lo[t] = f2bf(v - bf2f(h));
}

// ======================= degree / norm =======================
__global__ void k_deg_init(float* deg) {
  int n = blockIdx.x * blockDim.x + threadIdx.x;
  if (n < NN) deg[n] = 1.0f;                  // self loop
}
__global__ void k_deg_accum(const int* __restrict__ dst, float* __restrict__ deg) {
  int e = blockIdx.x * blockDim.x + threadIdx.x;
  if (e < EE) atomAddF(&deg[dst[e]], 1.0f);
}
__global__ void k_dinv(const float* __restrict__ deg, float* __restrict__ dinv) {
  int n = blockIdx.x * blockDim.x + threadIdx.x;
  if (n < NN) dinv[n] = rsqrtf(deg[n]);
}

// ======================= bf16x3 WMMA GEMM (double-buffered async) =======================
// C[M x Nc] = A[M x K] * B[K x Nc]; A row-major bf16 hi/lo, B pre-tiled bf16 hi/lo.
// M=NN, K=Nc=1024 compile-time. Block tile 64x128, 8 waves, wave tile 32x32,
// 2x2 WMMA x3 split terms, K-step 32, 2-deep LDS pipeline over ASYNCcnt.
#define BM 64
#define BN 128
#define BK 32
#define A_LD 40     // padded LDS stride (bf16 elems), multiple of 8 -> 16B aligned chunks
#define GK 1024
#define GN 1024

__device__ __forceinline__ bf16x16 fragA(const unsigned short* As, int m, int half) {
  FragPair p;
  int kb = half * 8;
  p.a = *(const u32x4*)(As + m * A_LD + kb);        // K = kb .. kb+7
  p.b = *(const u32x4*)(As + m * A_LD + 16 + kb);   // K = 16+kb .. 16+kb+7
  return __builtin_bit_cast(bf16x16, p);
}
__device__ __forceinline__ bf16x16 fragB(const unsigned short* Bs, int ntL, int lane) {
  const unsigned short* p = Bs + (((ntL << 5) + lane) << 4);  // 32B per lane, contiguous
  FragPair q;
  q.a = *(const u32x4*)p;
  q.b = *(const u32x4*)(p + 8);
  return __builtin_bit_cast(bf16x16, q);
}

__global__ __launch_bounds__(256) void gemm_bf16x3(
    const unsigned short* __restrict__ Ah, const unsigned short* __restrict__ Al,
    const unsigned short* __restrict__ Bth, const unsigned short* __restrict__ Btl,
    float* __restrict__ C) {
  __shared__ __align__(16) unsigned short Ash[2][BM * A_LD];
  __shared__ __align__(16) unsigned short Asl[2][BM * A_LD];
  __shared__ __align__(16) unsigned short Bsh[2][(BN / 16) * 32 * 16];  // 8KB each
  __shared__ __align__(16) unsigned short Bsl[2][(BN / 16) * 32 * 16];

  const int tid  = threadIdx.x;
  const int lane = tid & 31;
  const int wid  = tid >> 5;                // 0..7
  const int wm   = (wid >> 2) * 32;         // 0,32
  const int wn   = (wid & 3) * 32;          // 0,32,64,96
  const int blockM = blockIdx.y * BM;
  const int blockN = blockIdx.x * BN;
  const int half = lane >> 4;
  const int l15  = lane & 15;
  const int ntBase = blockN >> 4;

  // staging addresses (uniform per thread across K-steps)
  const int arow = tid >> 2;
  const int akc  = (tid & 3) * 8;
  int gr = blockM + arow; if (gr > NN - 1) gr = NN - 1;   // clamp: no EXEC divergence
  const size_t arowOff = (size_t)gr * GK + akc;

  // issue 6 async b128 loads for K-tile at k0 into LDS buffer `buf`
  auto stage = [&](int k0, int buf) {
    size_t gi = arowOff + k0;
    cp16(Ah + gi, Ash[buf] + arow * A_LD + akc);
    cp16(Al + gi, Asl[buf] + arow * A_LD + akc);
    size_t be = ((size_t)(k0 >> 5) * (GN >> 4) + ntBase) * 512 + (size_t)tid * 16;
    cp16(Bth + be,     Bsh[buf] + tid * 16);
    cp16(Bth + be + 8, Bsh[buf] + tid * 16 + 8);
    cp16(Btl + be,     Bsl[buf] + tid * 16);
    cp16(Btl + be + 8, Bsl[buf] + tid * 16 + 8);
  };

  f32x8 acc[2][2] = {};

  stage(0, 0);                               // prologue: tile 0 in flight
  for (int k0 = 0; k0 < GK; k0 += BK) {
    const int cur = (k0 >> 5) & 1;
    if (k0 + BK < GK) {
      stage(k0 + BK, cur ^ 1);               // tile t+1 in flight behind compute
      cp_wait_n<6>();                        // in-order ASYNCcnt: tile t complete
    } else {
      cp_wait_n<0>();
    }
    __syncthreads();

    bf16x16 AH[2], AL[2], BH[2], BL[2];
    #pragma unroll
    for (int tm = 0; tm < 2; ++tm) {
      AH[tm] = fragA(Ash[cur], wm + tm * 16 + l15, half);
      AL[tm] = fragA(Asl[cur], wm + tm * 16 + l15, half);
    }
    #pragma unroll
    for (int tn = 0; tn < 2; ++tn) {
      int ntL = (wn >> 4) + tn;              // 0..7 within block
      BH[tn] = fragB(Bsh[cur], ntL, lane);
      BL[tn] = fragB(Bsl[cur], ntL, lane);
    }

    #pragma unroll
    for (int tm = 0; tm < 2; ++tm) {
      #pragma unroll
      for (int tn = 0; tn < 2; ++tn) {
        acc[tm][tn] = __builtin_amdgcn_wmma_f32_16x16x32_bf16(
            false, AH[tm], false, BH[tn], (short)0, acc[tm][tn], false, false);
        acc[tm][tn] = __builtin_amdgcn_wmma_f32_16x16x32_bf16(
            false, AH[tm], false, BL[tn], (short)0, acc[tm][tn], false, false);
        acc[tm][tn] = __builtin_amdgcn_wmma_f32_16x16x32_bf16(
            false, AL[tm], false, BH[tn], (short)0, acc[tm][tn], false, false);
      }
    }
    __syncthreads();                         // all waves done reading buf `cur`
  }

  // ---- epilogue: C/D layout: elem r -> row r + half*8, col l15 ----
  #pragma unroll
  for (int tm = 0; tm < 2; ++tm) {
    #pragma unroll
    for (int tn = 0; tn < 2; ++tn) {
      int col = blockN + wn + tn * 16 + l15;
      #pragma unroll
      for (int r = 0; r < 8; ++r) {
        int row = blockM + wm + tm * 16 + r + half * 8;
        if (row < NN) C[(size_t)row * GN + col] = acc[tm][tn][r];
      }
    }
  }
}

// ======================= aggregation =======================
// self-loop term also serves as initializer: agg[n] = pre[n] * dinv[n]^2
__global__ void k_selfloop_init(const float* __restrict__ pre, const float* __restrict__ dinv,
                                float* __restrict__ agg) {
  size_t i = (size_t)blockIdx.x * blockDim.x + threadIdx.x;   // N*HH
  int n = (int)(i >> 10);
  float w = dinv[n];
  agg[i] = pre[i] * w * w;
}

__global__ void k_edge_agg(const float* __restrict__ pre,
                           const int* __restrict__ src, const int* __restrict__ dst,
                           const float* __restrict__ dinv, float* __restrict__ agg) {
  int e = blockIdx.x;
  int c = blockIdx.y * blockDim.x + threadIdx.x;
  int s = src[e], d = dst[e];
  float w = dinv[s] * dinv[d];
  atomAddF(&agg[(size_t)d * HH + c], pre[(size_t)s * HH + c] * w);
}

// h = leaky_relu(agg + b1) -> bf16 hi/lo (linear row-major layout)
__global__ void k_bias_lrelu_split(const float* __restrict__ agg, const float* __restrict__ b1,
                                   unsigned short* __restrict__ hi, unsigned short* __restrict__ lo) {
  size_t i = (size_t)blockIdx.x * blockDim.x + threadIdx.x;
  int c = (int)(i & (HH - 1));
  float v = agg[i] + b1[c];
  v = (v >= 0.f) ? v : NEG * v;
  unsigned short h = f2bf(v);
  hi[i] = h;
  lo[i] = f2bf(v - bf2f(h));
}

// final: mu = agg[:, :512]+bm -> out[z], out[mu]; logstd = agg[:, 512:]+bs -> out[logstd]
__global__ void k_final(const float* __restrict__ agg,
                        const float* __restrict__ bm, const float* __restrict__ bs,
                        float* __restrict__ out) {
  size_t i = (size_t)blockIdx.x * blockDim.x + threadIdx.x;   // N*HH
  int c = (int)(i & 1023);
  size_t n = i >> 10;
  float v = agg[i];
  const size_t blk = (size_t)NN * COUT;
  if (c < COUT) {
    float m = v + bm[c];
    out[n * COUT + c]       = m;   // z
    out[blk + n * COUT + c] = m;   // mu
  } else {
    out[2 * blk + n * COUT + (c - COUT)] = v + bs[c - COUT];  // logstd
  }
}

// ======================= host =======================
extern "C" void kernel_launch(void* const* d_in, const int* in_sizes, int n_in,
                              void* d_out, int out_size, void* d_ws, size_t ws_size,
                              hipStream_t stream) {
  const float* x   = (const float*)d_in[0];
  const int*   ei  = (const int*)d_in[1];     // jax default x64-off: int32, [2,E] flat
  const float* W1  = (const float*)d_in[2];
  const float* b1  = (const float*)d_in[3];
  const float* Wm  = (const float*)d_in[4];
  const float* bm  = (const float*)d_in[5];
  const float* Ws  = (const float*)d_in[6];
  const float* bs  = (const float*)d_in[7];
  const float* gnw = (const float*)d_in[8];
  const float* gnb = (const float*)d_in[9];
  const float* gms = (const float*)d_in[10];
  float* out = (float*)d_out;
  char* ws = (char*)d_ws;

  const int* srcI = ei;
  const int* dstI = ei + EE;

  // workspace carve-up (256B aligned)
  size_t off = 0;
  auto alloc = [&](size_t bytes) { off = (off + 255) & ~(size_t)255; size_t o = off; off += bytes; return o; };
  float* alpha  = (float*)(ws + alloc(CIN * 4));
  float* beta   = (float*)(ws + alloc(CIN * 4));
  float* csum   = (float*)(ws + alloc(CIN * 4));
  float* csum2  = (float*)(ws + alloc(CIN * 4));
  float* deg    = (float*)(ws + alloc(NN * 4));
  float* dinv   = (float*)(ws + alloc(NN * 4));
  unsigned short* xn_hi = (unsigned short*)(ws + alloc((size_t)NN * CIN * 2));
  unsigned short* xn_lo = (unsigned short*)(ws + alloc((size_t)NN * CIN * 2));
  unsigned short* w1_hi = (unsigned short*)(ws + alloc((size_t)CIN * HH * 2));
  unsigned short* w1_lo = (unsigned short*)(ws + alloc((size_t)CIN * HH * 2));
  unsigned short* wc_hi = (unsigned short*)(ws + alloc((size_t)HH * 1024 * 2));
  unsigned short* wc_lo = (unsigned short*)(ws + alloc((size_t)HH * 1024 * 2));
  unsigned short* h_hi  = (unsigned short*)(ws + alloc((size_t)NN * HH * 2));
  unsigned short* h_lo  = (unsigned short*)(ws + alloc((size_t)NN * HH * 2));
  float* pre    = (float*)(ws + alloc((size_t)NN * HH * 4));
  float* agg    = (float*)(ws + alloc((size_t)NN * HH * 4));
  (void)in_sizes; (void)n_in; (void)out_size; (void)ws_size;

  const int T = 256;
  const int rowsPer = (NN + 31) / 32;   // 313

  // --- GraphNorm ---
  k_stats_zero<<<CIN / T, T, 0, stream>>>(csum, csum2);
  k_stats_partial<<<dim3(CIN / T, 32), T, 0, stream>>>(x, csum, csum2, rowsPer);
  k_stats_final<<<CIN / T, T, 0, stream>>>(csum, csum2, gms, gnw, gnb, alpha, beta);
  k_norm_split<<<(int)(((size_t)NN * CIN) / T), T, 0, stream>>>(x, alpha, beta, xn_hi, xn_lo);

  // --- weights (pre-tiled into WMMA B-fragment layout) ---
  k_tile_w1<<<(int)(((size_t)CIN * HH) / T), T, 0, stream>>>(W1, w1_hi, w1_lo);
  k_tile_wcat<<<(int)(((size_t)HH * 1024) / T), T, 0, stream>>>(Wm, Ws, wc_hi, wc_lo);

  // --- degree normalization ---
  k_deg_init<<<(NN + T - 1) / T, T, 0, stream>>>(deg);
  k_deg_accum<<<(EE + T - 1) / T, T, 0, stream>>>(dstI, deg);
  k_dinv<<<(NN + T - 1) / T, T, 0, stream>>>(deg, dinv);

  dim3 ggrid(GN / BN, (NN + BM - 1) / BM);   // (8, 157)

  // --- conv1 ---
  gemm_bf16x3<<<ggrid, T, 0, stream>>>(xn_hi, xn_lo, w1_hi, w1_lo, pre);
  k_selfloop_init<<<(int)(((size_t)NN * HH) / T), T, 0, stream>>>(pre, dinv, agg);
  k_edge_agg<<<dim3(EE, HH / T), T, 0, stream>>>(pre, srcI, dstI, dinv, agg);
  k_bias_lrelu_split<<<(int)(((size_t)NN * HH) / T), T, 0, stream>>>(agg, b1, h_hi, h_lo);

  // --- conv_mu + conv_logstd (fused via packed weights) ---
  gemm_bf16x3<<<ggrid, T, 0, stream>>>(h_hi, h_lo, wc_hi, wc_lo, pre);
  k_selfloop_init<<<(int)(((size_t)NN * HH) / T), T, 0, stream>>>(pre, dinv, agg);
  k_edge_agg<<<dim3(EE, HH / T), T, 0, stream>>>(pre, srcI, dstI, dinv, agg);
  k_final<<<(int)(((size_t)NN * HH) / T), T, 0, stream>>>(agg, bm, bs, out);
}
